// CrossLinearAttentionBlock_29738353557765
// MI455X (gfx1250) — matrix operand order repaired and checked by reading, not compiled, not run on hardware
//
#include <hip/hip_runtime.h>

#define C_DIM 128
#define N_PIX 16384
#define SUB   64
#define NSUB  8
#define LDA_BF 136   // bf16 row stride for A-layout tiles / kvT (pad -> conflict-free b128)
#define LDA_W  68    // same, in dwords
#define LDK_BF 80    // kT/vT row stride in bf16 (64 data + pad, dword4-aligned)
#define LDK_W  40

typedef __attribute__((ext_vector_type(16))) __bf16 v16bf;
typedef __attribute__((ext_vector_type(8)))  float  v8f;

union FragU { uint4 q[2]; v16bf v; };

// native bf16 conversion (lets the backend use packed cvt instead of integer bit tricks)
__device__ __forceinline__ unsigned short f2bf(float f) {
  __bf16 b = (__bf16)f;
  return __builtin_bit_cast(unsigned short, b);
}
__device__ __forceinline__ float bf2f(unsigned short h) {
  return __uint_as_float(((unsigned)h) << 16);
}
__device__ __forceinline__ unsigned pack2(float a, float b) {
  return (unsigned)f2bf(a) | ((unsigned)f2bf(b) << 16);
}

// async global -> LDS (CDNA5 path, ASYNCcnt-tracked, no VGPR staging)
__device__ __forceinline__ void async_g2l_b128(void* lds_ptr, const void* gptr) {
  unsigned lds_off = (unsigned)(uintptr_t)lds_ptr;            // flat addr [31:0] == LDS offset
  unsigned long long ga = (unsigned long long)(uintptr_t)gptr;
  asm volatile("global_load_async_to_lds_b128 %0, %1, off"
               :: "v"(lds_off), "v"(ga) : "memory");
}
__device__ __forceinline__ void wait_async0() {
  asm volatile("s_wait_asynccnt 0x0" ::: "memory");
}

// A operand (16x32 bf16): lane m=lane&15, h=lane>>4; VGPR0-3 = words kb+4h..+3, VGPR4-7 = words kb+8+4h..+3
__device__ __forceinline__ v16bf fragA(const unsigned short* buf, int row, int rowWords, int kk, int h) {
  const unsigned* rw = (const unsigned*)buf + row * rowWords + kk * 16;
  FragU u;
  u.q[0] = *(const uint4*)(rw + 4 * h);
  u.q[1] = *(const uint4*)(rw + 8 + 4 * h);
  return u.v;
}
// B operand (32x16 bf16): lane n=lane&15, h=lane>>4; VGPR0-3 = words kb+4h..+3, VGPR4-7 = words kb+8+4h..+3
__device__ __forceinline__ v16bf fragB(const unsigned short* buf, int row, int rowWords, int kk, int h) {
  const unsigned* rw = (const unsigned*)buf + row * rowWords + kk * 16;
  FragU u;
  u.q[0] = *(const uint4*)(rw + 4 * h);
  u.q[1] = *(const uint4*)(rw + 8 + 4 * h);
  return u.v;
}

__device__ __forceinline__ v8f wmma_bf16(v16bf a, v16bf b, v8f c) {
  return __builtin_amdgcn_wmma_f32_16x16x32_bf16(false, a, false, b, (short)0, c, false, false);
}

// ---------------- kernel 0: weight transpose->bf16, zero accumulators ----------------
__global__ __launch_bounds__(256) void k_prep(
    const float* __restrict__ Wq, const float* __restrict__ Wk,
    const float* __restrict__ Wv, const float* __restrict__ Wo,
    unsigned short* __restrict__ WqT, unsigned short* __restrict__ WkT,
    unsigned short* __restrict__ WvT, unsigned short* __restrict__ WoT,
    float* __restrict__ KV, float* __restrict__ Ksum)
{
  int idx = blockIdx.x * 256 + threadIdx.x;
  if (idx < 4 * C_DIM * C_DIM) {
    int mtx = idx >> 14, e = idx & 16383;
    int k = e >> 7, j = e & 127;
    const float* W = (mtx == 0) ? Wq : (mtx == 1) ? Wk : (mtx == 2) ? Wv : Wo;
    unsigned short* WT = (mtx == 0) ? WqT : (mtx == 1) ? WkT : (mtx == 2) ? WvT : WoT;
    WT[j * C_DIM + k] = f2bf(W[e]);
  } else if (idx < 65536 + 8 * C_DIM * C_DIM) {
    KV[idx - 65536] = 0.f;
  } else if (idx < 65536 + 8 * C_DIM * C_DIM + 8 * C_DIM) {
    Ksum[idx - 65536 - 8 * C_DIM * C_DIM] = 0.f;
  }
}

// ---------------- kernel 1: layernorm(kv) -> K,V -> accumulate KV, Ksum ----------------
__global__ __launch_bounds__(256) void k_phase1(
    const float* __restrict__ kv_feat,
    const float* __restrict__ lnw_g, const float* __restrict__ lnb_g,
    const unsigned short* __restrict__ WkT, const unsigned short* __restrict__ WvT,
    float* __restrict__ KV, float* __restrict__ Ksum)
{
  __shared__ __align__(16) unsigned char regA[40960];           // sRaw f32[128][64] | kT+vT bf16[128][80]x2
  __shared__ __align__(16) unsigned short sA[SUB * LDA_BF];     // normalized kv, A-layout [n][c]
  __shared__ float lnw[C_DIM], lnb[C_DIM];
  __shared__ float mu[SUB], rs[SUB];

  float* sRaw = (float*)regA;
  unsigned short* kT = (unsigned short*)regA;                   // [c][n], stride LDK_BF
  unsigned short* vT = (unsigned short*)(regA + C_DIM * LDK_BF * 2);

  const int t = threadIdx.x;
  const int w = t >> 5, lane = t & 31;
  const int r16 = lane & 15, h = lane >> 4;
  const int m = w & 3, jg = (w >> 2) << 2;
  const int b = blockIdx.y;
  const int n_base = blockIdx.x * (SUB * NSUB);

  if (t < C_DIM) { lnw[t] = lnw_g[t]; lnb[t] = lnb_g[t]; }

  const v8f zero = {0.f,0.f,0.f,0.f,0.f,0.f,0.f,0.f};
  v8f accKV[8];
  #pragma unroll
  for (int j = 0; j < 8; ++j) accKV[j] = zero;
  float ksum_loc = 0.f;

  for (int s = 0; s < NSUB; ++s) {
    const int n0 = n_base + s * SUB;
    __syncthreads();                                            // protect regA/sA reuse
    // async load raw [c][n] tile (global -> LDS, no VGPR staging)
    #pragma unroll
    for (int i = 0; i < 8; ++i) {
      int e4 = t + 256 * i;                                     // 2048 float4s
      int row = e4 >> 4, col = e4 & 15;
      const float4* src = (const float4*)(kv_feat + ((size_t)(b * C_DIM + row)) * N_PIX + n0);
      async_g2l_b128(&((float4*)sRaw)[row * 16 + col], &src[col]);
    }
    wait_async0();
    __syncthreads();
    if (t < SUB) {                                              // per-pixel LN stats (f32)
      float s1 = 0.f, s2 = 0.f;
      for (int c = 0; c < C_DIM; ++c) { float x = sRaw[c * SUB + t]; s1 += x; s2 += x * x; }
      float mean = s1 * (1.f / C_DIM);
      float var  = s2 * (1.f / C_DIM) - mean * mean;
      mu[t] = mean; rs[t] = rsqrtf(var + 1e-5f);
    }
    __syncthreads();
    #pragma unroll
    for (int i = 0; i < 32; ++i) {                              // normalize -> bf16 A-layout [n][c]
      int e = t + 256 * i;
      int c = e >> 6, n = e & 63;
      float x = sRaw[e];
      sA[n * LDA_BF + c] = f2bf((x - mu[n]) * rs[n] * lnw[c] + lnb[c]);
    }
    __syncthreads();
    // K = elu(xn @ Wk)+1 and V = xn @ Wv ; write transposed [c][n] for the K^T V GEMM
    #pragma unroll
    for (int jt = 0; jt < 4; ++jt) {
      int j = jg + jt;
      v8f acck = zero, accv = zero;
      #pragma unroll
      for (int kk = 0; kk < 4; ++kk) {
        v16bf a = fragA(sA, m * 16 + r16, LDA_W, kk, h);
        acck = wmma_bf16(a, fragB(WkT, j * 16 + r16, 64, kk, h), acck);
        accv = wmma_bf16(a, fragB(WvT, j * 16 + r16, 64, kk, h), accv);
      }
      float ek[8];
      #pragma unroll
      for (int r = 0; r < 8; ++r) { float x = acck[r]; ek[r] = x > 0.f ? x + 1.f : __expf(x); }
      int cc = j * 16 + r16, nb = m * 16 + 8 * h;
      uint4 pk  = { pack2(ek[0],ek[1]),  pack2(ek[2],ek[3]),  pack2(ek[4],ek[5]),  pack2(ek[6],ek[7]) };
      uint4 pkv = { pack2(accv[0],accv[1]), pack2(accv[2],accv[3]),
                    pack2(accv[4],accv[5]), pack2(accv[6],accv[7]) };
      *(uint4*)(kT + cc * LDK_BF + nb) = pk;
      *(uint4*)(vT + cc * LDK_BF + nb) = pkv;
    }
    __syncthreads();
    // KV[c][d] += K^T V over this sub-tile (K-dim = 64 pixels -> 2 k-steps)
    #pragma unroll
    for (int j = 0; j < 8; ++j) {
      #pragma unroll
      for (int kk = 0; kk < 2; ++kk)
        accKV[j] = wmma_bf16(fragA(kT, w * 16 + r16, LDK_W, kk, h),
                             fragB(vT, j * 16 + r16, LDK_W, kk, h), accKV[j]);
    }
    if (t < C_DIM) {                                            // Ksum partial from kT
      for (int n = 0; n < SUB; ++n) ksum_loc += bf2f(kT[t * LDK_BF + n]);
    }
  }
  // flush accumulators
  #pragma unroll
  for (int j = 0; j < 8; ++j) {
    int d = j * 16 + r16;
    #pragma unroll
    for (int r = 0; r < 8; ++r) {
      int c = w * 16 + 8 * h + r;
      atomicAdd(&KV[(((size_t)b * C_DIM) + c) * C_DIM + d], accKV[j][r]);
    }
  }
  if (t < C_DIM) atomicAdd(&Ksum[b * C_DIM + t], ksum_loc);
}

// ---------------- kernel 2: layernorm(q) -> Q -> S=Q@KV / (Q.Ksum) -> @Wo + bo -> transposed out ----------------
__global__ __launch_bounds__(256) void k_phase2(
    const float* __restrict__ q_feat,
    const float* __restrict__ lnw_g, const float* __restrict__ lnb_g,
    const unsigned short* __restrict__ WqT, const unsigned short* __restrict__ WoT,
    const float* __restrict__ KV, const float* __restrict__ Ksum,
    const float* __restrict__ bo, float* __restrict__ out)
{
  __shared__ __align__(16) unsigned char regA[34816];           // sRaw f32[128][64] | kvT bf16[128][LDA_BF]
  __shared__ __align__(16) unsigned short qA[SUB * LDA_BF];     // q_norm -> Q -> S_scaled (sequential reuse)
  __shared__ float lnw[C_DIM], lnb[C_DIM], mu[SUB], rs[SUB], denom[SUB], rden[SUB];

  float* sRaw = (float*)regA;
  unsigned short* kvT = (unsigned short*)regA;                  // [d][c] B-layout

  const int t = threadIdx.x;
  const int w = t >> 5, lane = t & 31;
  const int r16 = lane & 15, h = lane >> 4;
  const int m = w & 3, jg = (w >> 2) << 2;
  const int b = blockIdx.y;
  const int n0 = blockIdx.x * SUB;
  const v8f zero = {0.f,0.f,0.f,0.f,0.f,0.f,0.f,0.f};

  if (t < C_DIM) { lnw[t] = lnw_g[t]; lnb[t] = lnb_g[t]; }
  #pragma unroll
  for (int i = 0; i < 8; ++i) {
    int e4 = t + 256 * i;
    int row = e4 >> 4, col = e4 & 15;
    const float4* src = (const float4*)(q_feat + ((size_t)(b * C_DIM + row)) * N_PIX + n0);
    async_g2l_b128(&((float4*)sRaw)[row * 16 + col], &src[col]);
  }
  wait_async0();
  __syncthreads();
  if (t < SUB) {
    float s1 = 0.f, s2 = 0.f;
    for (int c = 0; c < C_DIM; ++c) { float x = sRaw[c * SUB + t]; s1 += x; s2 += x * x; }
    float mean = s1 * (1.f / C_DIM);
    float var  = s2 * (1.f / C_DIM) - mean * mean;
    mu[t] = mean; rs[t] = rsqrtf(var + 1e-5f);
  }
  __syncthreads();
  #pragma unroll
  for (int i = 0; i < 32; ++i) {
    int e = t + 256 * i;
    int c = e >> 6, n = e & 63;
    float x = sRaw[e];
    qA[n * LDA_BF + c] = f2bf((x - mu[n]) * rs[n] * lnw[c] + lnb[c]);
  }
  __syncthreads();
  // KV -> bf16 B-layout (overwrites sRaw region)
  #pragma unroll 4
  for (int i = 0; i < 64; ++i) {
    int e = t + 256 * i;
    int c = e >> 7, d = e & 127;
    kvT[d * LDA_BF + c] = f2bf(KV[(((size_t)b * C_DIM) + c) * C_DIM + d]);
  }
  if (t < SUB) denom[t] = 0.f;
  __syncthreads();

  // Q projection + per-row denominator partials
  v8f qt[4];
  float part[8] = {0.f,0.f,0.f,0.f,0.f,0.f,0.f,0.f};
  #pragma unroll
  for (int jt = 0; jt < 4; ++jt) {
    int j = jg + jt;
    v8f acc = zero;
    #pragma unroll
    for (int kk = 0; kk < 4; ++kk)
      acc = wmma_bf16(fragA(qA, m * 16 + r16, LDA_W, kk, h),
                      fragB(WqT, j * 16 + r16, 64, kk, h), acc);
    #pragma unroll
    for (int r = 0; r < 8; ++r) { float x = acc[r]; acc[r] = x > 0.f ? x + 1.f : __expf(x); }
    qt[jt] = acc;
    float ks = Ksum[b * C_DIM + j * 16 + r16];
    #pragma unroll
    for (int r = 0; r < 8; ++r) part[r] += acc[r] * ks;
  }
  __syncthreads();                                              // all reads of q_norm done
  #pragma unroll
  for (int jt = 0; jt < 4; ++jt) {
    int cc = (jg + jt) * 16 + r16;
    #pragma unroll
    for (int r = 0; r < 8; ++r) qA[(m * 16 + 8 * h + r) * LDA_BF + cc] = f2bf(qt[jt][r]);
  }
  #pragma unroll
  for (int r = 0; r < 8; ++r) atomicAdd(&denom[m * 16 + 8 * h + r], part[r]);
  __syncthreads();
  if (t < SUB) rden[t] = 1.f / fmaxf(denom[t], 1e-6f);
  __syncthreads();

  // S = Q @ KV
  v8f st[4];
  #pragma unroll
  for (int jt = 0; jt < 4; ++jt) {
    int j = jg + jt;
    v8f acc = zero;
    #pragma unroll
    for (int kk = 0; kk < 4; ++kk)
      acc = wmma_bf16(fragA(qA, m * 16 + r16, LDA_W, kk, h),
                      fragB(kvT, j * 16 + r16, LDA_W, kk, h), acc);
    st[jt] = acc;
  }
  __syncthreads();                                              // all reads of Q done
  #pragma unroll
  for (int jt = 0; jt < 4; ++jt) {
    int cc = (jg + jt) * 16 + r16;
    #pragma unroll
    for (int r = 0; r < 8; ++r)
      qA[(m * 16 + 8 * h + r) * LDA_BF + cc] = f2bf(st[jt][r] * rden[m * 16 + 8 * h + r]);
  }
  __syncthreads();

  // out = S_scaled @ Wo + bo ; store transposed [c][n]
  #pragma unroll
  for (int jt = 0; jt < 4; ++jt) {
    int j = jg + jt;
    v8f acc = zero;
    #pragma unroll
    for (int kk = 0; kk < 4; ++kk)
      acc = wmma_bf16(fragA(qA, m * 16 + r16, LDA_W, kk, h),
                      fragB(WoT, j * 16 + r16, 64, kk, h), acc);
    int cc = j * 16 + r16;
    float bb = bo[cc];
    float* dst = out + (((size_t)b * C_DIM) + cc) * N_PIX + n0 + m * 16 + 8 * h;
    float4 f0 = { acc[0] + bb, acc[1] + bb, acc[2] + bb, acc[3] + bb };
    float4 f1 = { acc[4] + bb, acc[5] + bb, acc[6] + bb, acc[7] + bb };
    *(float4*)dst = f0;
    *(float4*)(dst + 4) = f1;
  }
}

extern "C" void kernel_launch(void* const* d_in, const int* in_sizes, int n_in,
                              void* d_out, int out_size, void* d_ws, size_t ws_size,
                              hipStream_t stream) {
  (void)in_sizes; (void)n_in; (void)out_size; (void)ws_size;
  const float* q_feat  = (const float*)d_in[0];
  const float* kv_feat = (const float*)d_in[1];
  const float* ln_q_w  = (const float*)d_in[2];
  const float* ln_q_b  = (const float*)d_in[3];
  const float* ln_kv_w = (const float*)d_in[4];
  const float* ln_kv_b = (const float*)d_in[5];
  const float* Wq = (const float*)d_in[6];
  const float* Wk = (const float*)d_in[7];
  const float* Wv = (const float*)d_in[8];
  const float* Wo = (const float*)d_in[9];
  const float* bo = (const float*)d_in[10];
  float* out = (float*)d_out;

  unsigned char* ws = (unsigned char*)d_ws;
  unsigned short* WqT = (unsigned short*)(ws + 0);
  unsigned short* WkT = (unsigned short*)(ws + 32768);
  unsigned short* WvT = (unsigned short*)(ws + 65536);
  unsigned short* WoT = (unsigned short*)(ws + 98304);
  float* KV   = (float*)(ws + 131072);                  // [B][C][C] f32, 512 KB
  float* Ksum = (float*)(ws + 131072 + 8 * 128 * 128 * 4);

  int prep_elems = 65536 + 131072 + 1024;
  k_prep<<<dim3((prep_elems + 255) / 256), 256, 0, stream>>>(Wq, Wk, Wv, Wo, WqT, WkT, WvT, WoT, KV, Ksum);
  k_phase1<<<dim3(N_PIX / (SUB * NSUB), 8), 256, 0, stream>>>(kv_feat, ln_kv_w, ln_kv_b, WkT, WvT, KV, Ksum);
  k_phase2<<<dim3(N_PIX / SUB, 8), 256, 0, stream>>>(q_feat, ln_q_w, ln_q_b, WqT, WoT, KV, Ksum, bo, out);
}
